// RGCN_57123065036980
// MI455X (gfx1250) — matrix operand order repaired and checked by reading, compiled, not run on hardware
//
#include <hip/hip_runtime.h>
#include <hip/hip_bf16.h>

// ---------------------------------------------------------------------------
// RGCN forward on MI455X (gfx1250, wave32, WMMA).
//
// Math rewrite (exactly matches reference):
//   tmp[n, r*64+i] = sum_{e: dst=n, type=r} x[src_e, i] * inv_deg[n]
//   x' = relu( [tmp | x] @ [W_stacked ; root] + bias )     (K = 512 + 64 = 576)
//   out[t] = dot(x[s_t], x[t_t])
// ---------------------------------------------------------------------------

typedef __attribute__((ext_vector_type(16))) _Float16 v16h;
typedef __attribute__((ext_vector_type(8)))  float    v8f;

#define RG_DIM    64
#define RG_NREL   8
#define RG_KTOT   576      // 8*64 relation blocks + 64 root block
#define RG_KSTAGE 288      // two LDS stages of the stacked weight (36 KB each)

// ---------------------------------------------------------------- fill -----
__global__ void rg_fill_f32(float* __restrict__ p, float v, long long n) {
    long long i = (long long)blockIdx.x * blockDim.x + threadIdx.x;
    if (i < n) p[i] = v;
}

__global__ void rg_fill_f32x4(float4* __restrict__ p, long long n4) {
    long long i = (long long)blockIdx.x * blockDim.x + threadIdx.x;
    if (i < n4) p[i] = make_float4(0.f, 0.f, 0.f, 0.f);
}

// ---------------------------------------------------------- degree ---------
__global__ void rg_deg_accum(const int* __restrict__ dst, float* __restrict__ deg,
                             int nedges) {
    int e = blockIdx.x * blockDim.x + threadIdx.x;
    if (e < nedges) atomicAdd(&deg[dst[e]], 1.0f);
}

__global__ void rg_deg_inv(float* __restrict__ deg, int n) {
    int i = blockIdx.x * blockDim.x + threadIdx.x;
    if (i < n) deg[i] = 1.0f / fmaxf(deg[i], 1.0f);
}

// ---------------------------------------------------------- gather x0 ------
__global__ void rg_gather_x(const int* __restrict__ x_idx,
                            const float* __restrict__ node_emb,
                            float* __restrict__ x, int nnodes) {
    long long t = (long long)blockIdx.x * blockDim.x + threadIdx.x;
    if (t >= (long long)nnodes * RG_DIM) return;
    int n = (int)(t >> 6), d = (int)(t & 63);
    x[t] = node_emb[(long long)x_idx[n] * RG_DIM + d];
}

// ----------------------------------------------------------- scatter -------
// 16 threads per edge, 4 floats each; inv_deg folded into the message.
__global__ void rg_scatter(const int* __restrict__ src, const int* __restrict__ dst,
                           const int* __restrict__ etype,
                           const float* __restrict__ x,
                           const float* __restrict__ invdeg,
                           float* __restrict__ tmp, int nedges) {
    long long t = (long long)blockIdx.x * blockDim.x + threadIdx.x;
    int e = (int)(t >> 4);
    int part = (int)(t & 15);
    if (e >= nedges) return;
    int s = src[e], d = dst[e], r = etype[e];
    float sc = invdeg[d];
    const float4 v = *(const float4*)(x + (long long)s * RG_DIM + part * 4);
    float* o = tmp + (long long)d * (RG_NREL * RG_DIM) + r * RG_DIM + part * 4;
    atomicAdd(o + 0, v.x * sc);
    atomicAdd(o + 1, v.y * sc);
    atomicAdd(o + 2, v.z * sc);
    atomicAdd(o + 3, v.w * sc);
}

// ------------------------------------------------------- WMMA GEMM ---------
// D[M=nodes, N=64] = relu( A[M, 576] x B[576, 64] + bias ), fp32 accumulate.
// A: k<512 -> tmp (already inv_deg scaled), k>=512 -> x (root block).
// B: stacked [W[r,i,o] ; root[i,o]] staged transposed ([o][k], f16) in LDS.
// Block = 256 threads = 8 waves = 2 M-tiles x 4 N-tiles. 18 WMMAs per wave.
__global__ __launch_bounds__(256)
void rg_layer_gemm(const float* __restrict__ tmp,   // [N, 512]
                   const float* __restrict__ x,     // [N, 64]
                   const float* __restrict__ Wl,    // [8, 64, 64]
                   const float* __restrict__ rootl, // [64, 64]
                   const float* __restrict__ biasl, // [64]
                   float* __restrict__ xout,        // [N, 64]
                   int nnodes) {
    __shared__ __align__(32) _Float16 wt[RG_DIM * RG_KSTAGE]; // 36864 B

    const int wave  = threadIdx.x >> 5;
    const int lane  = threadIdx.x & 31;
    const int row16 = lane & 15;     // A row / B-C column within the tile
    const int hi    = lane >> 4;     // K-half selector
    const int mtile = blockIdx.x * 32 + (wave >> 2) * 16;
    const int ntile = (wave & 3) * 16;
    const long long mrow = mtile + row16;

    // Prefetch this lane's A rows into cache (global_prefetch_b8).
    __builtin_prefetch(tmp + mrow * (RG_NREL * RG_DIM), 0, 0);
    __builtin_prefetch(x + mrow * RG_DIM, 0, 0);

    v8f acc = {};

    #pragma unroll
    for (int stage = 0; stage < 2; ++stage) {
        const int kbase = stage * RG_KSTAGE;
        if (stage) __syncthreads();  // previous stage fully consumed
        // Cooperative stage of transposed f16 weights: wt[o][k_local].
        for (int idx = threadIdx.x; idx < RG_DIM * RG_KSTAGE; idx += 256) {
            int o  = idx / RG_KSTAGE;
            int kl = idx - o * RG_KSTAGE;
            int k  = kbase + kl;
            float v = (k < RG_NREL * RG_DIM)
                          ? Wl[((k >> 6) << 12) + ((k & 63) << 6) + o]
                          : rootl[((k - RG_NREL * RG_DIM) << 6) + o];
            wt[idx] = (_Float16)v;
        }
        __syncthreads();

        #pragma unroll
        for (int kl = 0; kl < RG_KSTAGE; kl += 32) {
            const int k0 = kbase + kl;  // uniform across wave
            const float* asrc = (k0 < RG_NREL * RG_DIM)
                                    ? (tmp + mrow * (RG_NREL * RG_DIM) + k0)
                                    : (x + mrow * RG_DIM + (k0 - RG_NREL * RG_DIM));
            // A fragment 16x32 f16: halves 0-7 = K hi*8.., halves 8-15 = K 16+hi*8..
            v16h a;
            #pragma unroll
            for (int j = 0; j < 8; ++j) {
                a[j]     = (_Float16)asrc[hi * 8 + j];
                a[j + 8] = (_Float16)asrc[16 + hi * 8 + j];
            }
            // B fragment 32x16 f16: column ntile+row16, halves 0-7 = K hi*8.., halves 8-15 = K 16+hi*8..
            const _Float16* bs = &wt[(ntile + row16) * RG_KSTAGE + kl + hi * 8];
            v16h b;
            #pragma unroll
            for (int j = 0; j < 8; ++j) { b[j] = bs[j]; b[j + 8] = bs[16 + j]; }

            acc = __builtin_amdgcn_wmma_f32_16x16x32_f16(
                false, a, false, b, (short)0, acc, false, false);
        }
    }

    // Epilogue: C/D element (v, lane) -> M = v + 8*hi, N = lane&15.
    const int col = ntile + row16;
    const float bv = biasl[col];
    #pragma unroll
    for (int v = 0; v < 8; ++v) {
        int m = mtile + v + 8 * hi;
        if (m < nnodes) {
            float r = acc[v] + bv;
            xout[(long long)m * RG_DIM + col] = r > 0.0f ? r : 0.0f;
        }
    }
}

// ------------------------------------------------------------- score -------
// One wave per target edge; 2 floats per lane; butterfly reduce in wave32.
__global__ void rg_score(const int* __restrict__ ts, const int* __restrict__ tt,
                         const float* __restrict__ x, float* __restrict__ out,
                         int ntgt) {
    int w = (int)(((long long)blockIdx.x * blockDim.x + threadIdx.x) >> 5);
    int lane = threadIdx.x & 31;
    if (w >= ntgt) return;  // whole-wave uniform exit
    long long s = ts[w], t = tt[w];
    float2 a = *(const float2*)(x + s * RG_DIM + lane * 2);
    float2 b = *(const float2*)(x + t * RG_DIM + lane * 2);
    float p = a.x * b.x + a.y * b.y;
    #pragma unroll
    for (int off = 16; off > 0; off >>= 1) p += __shfl_xor(p, off, 32);
    if (lane == 0) out[w] = p;
}

// ------------------------------------------------------------- launch ------
extern "C" void kernel_launch(void* const* d_in, const int* in_sizes, int n_in,
                              void* d_out, int out_size, void* d_ws, size_t ws_size,
                              hipStream_t stream) {
    const int*   x_idx    = (const int*)d_in[0];
    const int*   eidx     = (const int*)d_in[1];   // [2, E]
    const int*   etype    = (const int*)d_in[2];   // [E]
    const int*   tgt      = (const int*)d_in[3];   // [2, T]
    const float* node_emb = (const float*)d_in[4]; // [N, 64]
    const float* weight   = (const float*)d_in[5]; // [2, 8, 64, 64]
    const float* root     = (const float*)d_in[6]; // [2, 64, 64]
    const float* bias     = (const float*)d_in[7]; // [2, 64]
    float*       out      = (float*)d_out;

    const int nnodes = in_sizes[0];
    const int nedges = in_sizes[1] / 2;
    const int ntgt   = in_sizes[3] / 2;
    const int nlayers = in_sizes[7] / RG_DIM;

    const int* src = eidx;
    const int* dst = eidx + nedges;
    const int* ts  = tgt;
    const int* tt  = tgt + ntgt;

    // Workspace layout (float, 16B-aligned slices): deg | xa | xb | tmp
    long long npad = ((long long)nnodes + 3) & ~3LL;
    float* deg = (float*)d_ws;
    float* xa  = deg + npad;
    float* xb  = xa + npad * RG_DIM;
    float* tmp = xb + npad * RG_DIM;

    const int B = 256;

    // inverse mean-degree
    rg_fill_f32<<<(int)((nnodes + B - 1) / B), B, 0, stream>>>(deg, 0.0f, nnodes);
    rg_deg_accum<<<(nedges + B - 1) / B, B, 0, stream>>>(dst, deg, nedges);
    rg_deg_inv<<<(nnodes + B - 1) / B, B, 0, stream>>>(deg, nnodes);

    // x0 = node_emb[x_idx]
    {
        long long tot = (long long)nnodes * RG_DIM;
        rg_gather_x<<<(int)((tot + B - 1) / B), B, 0, stream>>>(x_idx, node_emb, xa, nnodes);
    }

    float* xin = xa;
    float* xout_ = xb;
    for (int l = 0; l < nlayers; ++l) {
        long long tmp4 = (long long)nnodes * (RG_NREL * RG_DIM) / 4;
        rg_fill_f32x4<<<(int)((tmp4 + B - 1) / B), B, 0, stream>>>((float4*)tmp, tmp4);

        long long sthreads = (long long)nedges * 16;
        rg_scatter<<<(int)((sthreads + B - 1) / B), B, 0, stream>>>(
            src, dst, etype, xin, deg, tmp, nedges);

        int gblocks = (nnodes + 31) / 32;  // 2 M-tiles of 16 per block
        rg_layer_gemm<<<gblocks, 256, 0, stream>>>(
            tmp, xin,
            weight + (long long)l * RG_NREL * RG_DIM * RG_DIM,
            root + (long long)l * RG_DIM * RG_DIM,
            bias + (long long)l * RG_DIM,
            xout_, nnodes);

        float* t2 = xin; xin = xout_; xout_ = t2;
    }

    // edge scores
    {
        long long wthreads = (long long)ntgt * 32;
        rg_score<<<(int)((wthreads + B - 1) / B), B, 0, stream>>>(ts, tt, xin, out, ntgt);
    }
}
